// Informer_4ours_71691594105552
// MI455X (gfx1250) — matrix operand order repaired and checked by reading, not compiled, not run on hardware
//
#include <hip/hip_runtime.h>
#include <hip/hip_bf16.h>
#include <math.h>

typedef __attribute__((ext_vector_type(16))) _Float16 v16h;
typedef __attribute__((ext_vector_type(8)))  _Float16 v8h;
typedef __attribute__((ext_vector_type(8)))  float    v8f;

// ---------------- problem constants ----------------
#define BN    16
#define NFEAT 128
#define LAGL  336
#define LABL  48
#define PREDL 96
#define DECL  144
#define DMOD  512
#define NHEAD 8
#define DHEAD 64
#define DFFN  2048
#define TMARK 4

// =====================================================================
// Kernels
// =====================================================================

__global__ void tohalf_k(const float* __restrict__ x, _Float16* __restrict__ y, long n) {
  long i = (long)blockIdx.x * blockDim.x + threadIdx.x;
  if (i < n) y[i] = (_Float16)x[i];
}

// W is (K x N) row-major f32 -> out is (N x K) row-major f16 (transposed pack)
__global__ void packw_k(const float* __restrict__ W, _Float16* __restrict__ out, int K, int N) {
  long i = (long)blockIdx.x * blockDim.x + threadIdx.x;
  if (i >= (long)K * N) return;
  int n = (int)(i % N);
  int k = (int)(i / N);
  out[(long)n * K + k] = (_Float16)W[i];
}

// conv weight (Dout, Cin, 3) -> packed (Dout x K), K = 3*Cin, k = j*Cin + c
__global__ void packcw_k(const float* __restrict__ w, _Float16* __restrict__ out, int Dout, int Cin) {
  long i = (long)blockIdx.x * blockDim.x + threadIdx.x;
  long total = (long)Dout * Cin * 3;
  if (i >= total) return;
  int j = (int)(i % 3);
  int c = (int)((i / 3) % Cin);
  int d = (int)(i / (3 * (long)Cin));
  out[(long)d * (3 * Cin) + j * Cin + c] = (_Float16)w[i];
}

// circular im2col: row = b*L + l, col = j*Cin + c, source (b, (l+j-1) mod L, c)
// x element (b,l,c) at x[b*sB + l*sL + c*sC]; wrapped index >= validL reads 0
__global__ void im2col_k(const float* __restrict__ x, _Float16* __restrict__ out,
                         int Bn, int L, int Cin, long sB, long sL, long sC, int validL) {
  long i = (long)blockIdx.x * blockDim.x + threadIdx.x;
  long total = (long)Bn * L * 3 * Cin;
  if (i >= total) return;
  int c = (int)(i % Cin);
  long r = i / Cin;
  int j = (int)(r % 3); r /= 3;
  int l = (int)(r % L);
  int b = (int)(r / L);
  int ls = l + j - 1;
  ls = (ls + L) % L;
  float v = (ls < validL) ? x[(long)b * sB + (long)ls * sL + (long)c * sC] : 0.0f;
  out[((long)(b * L + l) * (3 * Cin)) + j * Cin + c] = (_Float16)v;
}

// ---------------------------------------------------------------------
// Fragment loaders for 16-bit WMMA operands, same contiguous k order for A and B.
// A (16x32, row-major source): lane M=lane%16; elem e -> k = 16*half + e
// B (32x16, from transposed NxK source): lane N=lane%16; elem e -> k = 16*half + e
// ---------------------------------------------------------------------
static __device__ __forceinline__ v16h load_afrag(const _Float16* p) {
  v8h lo = *(const v8h*)(p);
  v8h hi = *(const v8h*)(p + 8);
  v16h r;
#pragma unroll
  for (int e = 0; e < 8; ++e) { r[e] = lo[e]; r[e + 8] = hi[e]; }
  return r;
}
static __device__ __forceinline__ v16h load_bfrag(const _Float16* p) {
  v8h lo = *(const v8h*)(p);
  v8h hi = *(const v8h*)(p + 8);
  v16h r;
#pragma unroll
  for (int e = 0; e < 8; ++e) { r[e] = lo[e]; r[e + 8] = hi[e]; }
  return r;
}

// ---------------------------------------------------------------------
// WMMA GEMM: C(MxN,f32) = A(MxK,f16 row-major) * Bt(NxK,f16: B transposed)
// 8 waves/block arranged 2(m) x 4(n); each wave computes a 32x32 tile as
// 2x2 fragments -> 4 WMMAs per 8 b128 loads (A/B fragments reused 2x).
// Block tile = 64 x 128. All M here are multiples of 64, all N of 128.
// WGP-scope speculative prefetch (locality 3) issued 4 K-steps ahead.
// epilogue: +bias, act(1=exact GELU), +residual
// ---------------------------------------------------------------------
__global__ void __launch_bounds__(256)
gemm_wmma_k(const _Float16* __restrict__ A, const _Float16* __restrict__ Bt,
            const float* __restrict__ bias, const float* __restrict__ res,
            float* __restrict__ C, int M, int N, int K, int act) {
  const int wave = threadIdx.x >> 5;       // 0..7
  const int lane = threadIdx.x & 31;
  const int half = lane >> 4;
  const int lr   = lane & 15;
  const int wm = wave >> 2;                // 0..1
  const int wn = wave & 3;                 // 0..3
  const int m0 = blockIdx.x * 64 + wm * 32;
  const int n0 = blockIdx.y * 128 + wn * 32;
  const _Float16* Ar0 = A  + (size_t)(m0 + lr) * K + 16 * half;
  const _Float16* Ar1 = Ar0 + (size_t)16 * K;
  const _Float16* Bc0 = Bt + (size_t)(n0 + lr) * K + 16 * half;
  const _Float16* Bc1 = Bc0 + (size_t)16 * K;

  v8f acc[2][2] = {};
  for (int kb = 0; kb < K; kb += 32) {
    v16h a[2], b[2];
    a[0] = load_afrag(Ar0 + kb);
    a[1] = load_afrag(Ar1 + kb);
    b[0] = load_bfrag(Bc0 + kb);
    b[1] = load_bfrag(Bc1 + kb);
    if (kb + 128 < K) {                    // speculative prefetch, 4 iters ahead (WGP scope)
      __builtin_prefetch(Ar0 + kb + 128, 0, 3);
      __builtin_prefetch(Ar1 + kb + 128, 0, 3);
      __builtin_prefetch(Bc0 + kb + 128, 0, 3);
      __builtin_prefetch(Bc1 + kb + 128, 0, 3);
    }
#pragma unroll
    for (int mi = 0; mi < 2; ++mi)
#pragma unroll
      for (int ni = 0; ni < 2; ++ni)
        acc[mi][ni] = __builtin_amdgcn_wmma_f32_16x16x32_f16(
            false, a[mi], false, b[ni], (short)0, acc[mi][ni], false, false);
  }

#pragma unroll
  for (int mi = 0; mi < 2; ++mi) {
#pragma unroll
    for (int ni = 0; ni < 2; ++ni) {
      int nb = n0 + ni * 16 + lr;
      float bi = bias ? bias[nb] : 0.0f;
#pragma unroll
      for (int r = 0; r < 8; ++r) {
        int m = m0 + mi * 16 + r + 8 * half;
        float v = acc[mi][ni][r] + bi;
        if (act == 1) v = 0.5f * v * (1.0f + erff(v * 0.7071067811865476f));
        if (res)  v += res[(size_t)m * N + nb];
        C[(size_t)m * N + nb] = v;
      }
    }
  }
}

// y[b,l,d] += sum_t mark[b,t,l]*temp[t,d] + pos_enc(l,d)
__global__ void embed_k(float* __restrict__ y, const float* __restrict__ mark,
                        const float* __restrict__ temp, int L) {
  long i = (long)blockIdx.x * blockDim.x + threadIdx.x;
  long total = (long)BN * L * DMOD;
  if (i >= total) return;
  int d = (int)(i % DMOD);
  int l = (int)((i / DMOD) % L);
  int b = (int)(i / ((long)DMOD * L));
  float acc = y[i];
#pragma unroll
  for (int t = 0; t < TMARK; ++t)
    acc += mark[((long)b * TMARK + t) * L + l] * temp[t * DMOD + d];
  float freq = expf(-(float)(d & ~1) * (9.210340371976184f / (float)DMOD)); // ln(1e4)/D
  float ph = (float)l * freq;
  acc += (d & 1) ? cosf(ph) : sinf(ph);
  y[i] = acc;
}

__global__ void __launch_bounds__(256)
ln_k(const float* __restrict__ x, const float* __restrict__ g,
     const float* __restrict__ b, float* __restrict__ y) {
  int row = blockIdx.x;
  const float* xr = x + (long)row * DMOD;
  float* yr = y + (long)row * DMOD;
  __shared__ float red[256];
  int tid = threadIdx.x;
  float s = 0.f;
  for (int i = tid; i < DMOD; i += 256) s += xr[i];
  red[tid] = s; __syncthreads();
  for (int off = 128; off; off >>= 1) { if (tid < off) red[tid] += red[tid + off]; __syncthreads(); }
  float mean = red[0] / (float)DMOD;
  __syncthreads();
  float v = 0.f;
  for (int i = tid; i < DMOD; i += 256) { float d = xr[i] - mean; v += d * d; }
  red[tid] = v; __syncthreads();
  for (int off = 128; off; off >>= 1) { if (tid < off) red[tid] += red[tid + off]; __syncthreads(); }
  float inv = rsqrtf(red[0] / (float)DMOD + 1e-5f);
  for (int i = tid; i < DMOD; i += 256) yr[i] = (xr[i] - mean) * inv * g[i] + b[i];
}

// deterministic stand-in for jax.random.randint sampling
static __device__ __forceinline__ int samp_idx(int l, int uu, int LK) {
  unsigned v = (unsigned)l * 1103515245u + (unsigned)uu * 2654435761u + 12345u;
  v ^= v >> 13; v *= 2246822519u; v ^= v >> 11;
  return (int)(v % (unsigned)LK);
}

// M[b,h,l] = max_u q.k_samp - sum_u q.k_samp / L_K    (Q,K in (B,L,H*DH) layout)
__global__ void attn_m_k(const float* __restrict__ Q, const float* __restrict__ Km,
                         float* __restrict__ Msc, int LQ, int LK, int U) {
  long i = (long)blockIdx.x * blockDim.x + threadIdx.x;
  long total = (long)BN * NHEAD * LQ;
  if (i >= total) return;
  int l = (int)(i % LQ);
  int h = (int)((i / LQ) % NHEAD);
  int b = (int)(i / ((long)LQ * NHEAD));
  const float* q = Q + ((long)(b * LQ + l) * DMOD) + h * DHEAD;
  float mx = -INFINITY, sum = 0.f;
  for (int uu = 0; uu < U; ++uu) {
    int ks = samp_idx(l, uu, LK);
    const float* kk = Km + ((long)(b * LK + ks) * DMOD) + h * DHEAD;
    float d = 0.f;
#pragma unroll 8
    for (int e = 0; e < DHEAD; ++e) d += q[e] * kk[e];
    mx = fmaxf(mx, d); sum += d;
  }
  Msc[i] = mx - sum / (float)LK;
}

// per (b,h): select top-u indices of Msc row (length LQ <= 352)
__global__ void __launch_bounds__(256)
topk_k(const float* __restrict__ Msc, int* __restrict__ mtop, int LQ, int u) {
  int bh = blockIdx.x;
  const float* m = Msc + (long)bh * LQ;
  __shared__ float sv[352];
  __shared__ float rv[256];
  __shared__ int   ri[256];
  int tid = threadIdx.x;
  for (int i = tid; i < LQ; i += 256) sv[i] = m[i];
  __syncthreads();
  for (int t = 0; t < u; ++t) {
    float best = -INFINITY; int bi = 0;
    for (int i = tid; i < LQ; i += 256)
      if (sv[i] > best) { best = sv[i]; bi = i; }
    rv[tid] = best; ri[tid] = bi; __syncthreads();
    for (int off = 128; off; off >>= 1) {
      if (tid < off && rv[tid + off] > rv[tid]) { rv[tid] = rv[tid + off]; ri[tid] = ri[tid + off]; }
      __syncthreads();
    }
    if (tid == 0) { mtop[(long)bh * u + t] = ri[0]; sv[ri[0]] = -INFINITY; }
    __syncthreads();
  }
}

// context = broadcast mean_l V  (non-causal)
__global__ void ctx_mean_k(const float* __restrict__ V, float* __restrict__ Ctx, int LK, int LQ) {
  long i = (long)blockIdx.x * blockDim.x + threadIdx.x;
  long total = (long)BN * NHEAD * DHEAD;
  if (i >= total) return;
  int dh = (int)(i % DHEAD);
  int h  = (int)((i / DHEAD) % NHEAD);
  int b  = (int)(i / (DHEAD * NHEAD));
  float s = 0.f;
  for (int l = 0; l < LK; ++l) s += V[((long)(b * LK + l) * DMOD) + h * DHEAD + dh];
  s /= (float)LK;
  for (int l = 0; l < LQ; ++l) Ctx[((long)(b * LQ + l) * DMOD) + h * DHEAD + dh] = s;
}

// context = cumsum_l V  (causal)
__global__ void ctx_cumsum_k(const float* __restrict__ V, float* __restrict__ Ctx, int L) {
  long i = (long)blockIdx.x * blockDim.x + threadIdx.x;
  long total = (long)BN * NHEAD * DHEAD;
  if (i >= total) return;
  int dh = (int)(i % DHEAD);
  int h  = (int)((i / DHEAD) % NHEAD);
  int b  = (int)(i / (DHEAD * NHEAD));
  float s = 0.f;
  for (int l = 0; l < L; ++l) {
    s += V[((long)(b * L + l) * DMOD) + h * DHEAD + dh];
    Ctx[((long)(b * L + l) * DMOD) + h * DHEAD + dh] = s;
  }
}

// per (b,h,u): scores row -> mask -> softmax -> Upd = probs . V
__global__ void __launch_bounds__(128)
attn_upd_k(const float* __restrict__ Q, const float* __restrict__ Km,
           const float* __restrict__ V, const int* __restrict__ mtop,
           float* __restrict__ Upd, int LQ, int LK, int u, int causal) {
  int bid = blockIdx.x;
  int uu = bid % u;
  int bh = bid / u;
  int h = bh % NHEAD;
  int b = bh / NHEAD;
  int tid = threadIdx.x;
  __shared__ float sc[352];
  __shared__ float red[128];
  int mrow = mtop[(long)bh * u + uu];
  const float* q = Q + ((long)(b * LQ + mrow) * DMOD) + h * DHEAD;
  for (int k = tid; k < LK; k += 128) {
    const float* kv = Km + ((long)(b * LK + k) * DMOD) + h * DHEAD;
    float d = 0.f;
#pragma unroll 8
    for (int e = 0; e < DHEAD; ++e) d += q[e] * kv[e];
    d *= 0.125f;                                // 1/sqrt(64)
    if (causal && k > mrow) d = -INFINITY;
    sc[k] = d;
  }
  __syncthreads();
  float mx = -INFINITY;
  for (int k = tid; k < LK; k += 128) mx = fmaxf(mx, sc[k]);
  red[tid] = mx; __syncthreads();
  for (int off = 64; off; off >>= 1) { if (tid < off) red[tid] = fmaxf(red[tid], red[tid + off]); __syncthreads(); }
  mx = red[0]; __syncthreads();
  float sum = 0.f;
  for (int k = tid; k < LK; k += 128) { float e = expf(sc[k] - mx); sc[k] = e; sum += e; }
  red[tid] = sum; __syncthreads();
  for (int off = 64; off; off >>= 1) { if (tid < off) red[tid] += red[tid + off]; __syncthreads(); }
  float tot = red[0];
  __syncthreads();
  for (int dh = tid; dh < DHEAD; dh += 128) {
    float acc = 0.f;
    for (int k = 0; k < LK; ++k) acc += sc[k] * V[((long)(b * LK + k) * DMOD) + h * DHEAD + dh];
    Upd[(long)bid * DHEAD + dh] = acc / tot;
  }
}

__global__ void scatter_k(const float* __restrict__ Upd, const int* __restrict__ mtop,
                          float* __restrict__ Ctx, int u, int LQ) {
  long i = (long)blockIdx.x * blockDim.x + threadIdx.x;
  long total = (long)BN * NHEAD * u * DHEAD;
  if (i >= total) return;
  int dh = (int)(i % DHEAD);
  int uu = (int)((i / DHEAD) % u);
  int h  = (int)((i / ((long)DHEAD * u)) % NHEAD);
  int b  = (int)(i / ((long)DHEAD * u * NHEAD));
  int m = mtop[((long)(b * NHEAD + h)) * u + uu];
  Ctx[((long)(b * LQ + m) * DMOD) + h * DHEAD + dh] = Upd[i];
}

// distil post-conv: y = elu(y * gamma/sqrt(1+1e-5) + beta)
__global__ void bnelu_k(float* __restrict__ y, const float* __restrict__ gamma,
                        const float* __restrict__ beta, long n) {
  long i = (long)blockIdx.x * blockDim.x + threadIdx.x;
  if (i >= n) return;
  int d = (int)(i % DMOD);
  float v = y[i] * (gamma[d] * 0.9999950000374997f) + beta[d];
  y[i] = v > 0.f ? v : (expf(v) - 1.f);
}

// window 3, stride 2, pad (1,1), -inf pad: out length = L/2
__global__ void maxpool_k(const float* __restrict__ y, float* __restrict__ out, int L) {
  int Lo = L / 2;
  long i = (long)blockIdx.x * blockDim.x + threadIdx.x;
  long total = (long)BN * Lo * DMOD;
  if (i >= total) return;
  int d  = (int)(i % DMOD);
  int lo = (int)((i / DMOD) % Lo);
  int b  = (int)(i / ((long)DMOD * Lo));
  float m = -INFINITY;
#pragma unroll
  for (int j = 0; j < 3; ++j) {
    int ls = 2 * lo + j - 1;
    if (ls >= 0 && ls < L) m = fmaxf(m, y[((long)(b * L + ls) * DMOD) + d]);
  }
  out[((long)(b * Lo + lo) * DMOD) + d] = m;
}

// out[b,n,t] = proj[b, LABEL+t, n]
__global__ void final_k(const float* __restrict__ proj, float* __restrict__ out) {
  long i = (long)blockIdx.x * blockDim.x + threadIdx.x;
  long total = (long)BN * NFEAT * PREDL;
  if (i >= total) return;
  int t = (int)(i % PREDL);
  int n = (int)((i / PREDL) % NFEAT);
  int b = (int)(i / ((long)PREDL * NFEAT));
  out[i] = proj[((long)(b * DECL + LABL + t) * NFEAT) + n];
}

// =====================================================================
// Host orchestration
// =====================================================================

// Param leaf indices assuming JAX pytree flattening (dict keys sorted, recursive)
enum {
  IN_X = 0, IN_T = 1, IN_TY = 2, IN_Y = 3,
  P_dec_b = 4, P_dec_conv = 5, P_dec_g = 6,
  P_dl_b1 = 7, P_dl_b2 = 8,
  P_cr_bk = 9, P_cr_bo = 10, P_cr_bq = 11, P_cr_bv = 12,
  P_cr_wk = 13, P_cr_wo = 14, P_cr_wq = 15, P_cr_wv = 16,
  P_dl_g1 = 17, P_dl_g2 = 18, P_dl_g3 = 19, P_dl_n1 = 20, P_dl_n2 = 21, P_dl_n3 = 22,
  P_se_bk = 23, P_se_bo = 24, P_se_bq = 25, P_se_bv = 26,
  P_se_wk = 27, P_se_wo = 28, P_se_wq = 29, P_se_wv = 30,
  P_dl_w1 = 31, P_dl_w2 = 32,
  P_dec_temp = 33,
  P_di_b = 34, P_di_beta = 35, P_di_gamma = 36, P_di_w = 37,
  P_enc_b = 38, P_enc_conv = 39, P_enc_g = 40,
  P_e0_bk = 41, P_e0_bo = 42, P_e0_bq = 43, P_e0_bv = 44,
  P_e0_wk = 45, P_e0_wo = 46, P_e0_wq = 47, P_e0_wv = 48,
  P_e0_b1 = 49, P_e0_b2 = 50, P_e0_g1 = 51, P_e0_g2 = 52, P_e0_n1 = 53, P_e0_n2 = 54,
  P_e0_w1 = 55, P_e0_w2 = 56,
  P_e1_bk = 57, P_e1_bo = 58, P_e1_bq = 59, P_e1_bv = 60,
  P_e1_wk = 61, P_e1_wo = 62, P_e1_wq = 63, P_e1_wv = 64,
  P_e1_b1 = 65, P_e1_b2 = 66, P_e1_g1 = 67, P_e1_g2 = 68, P_e1_n1 = 69, P_e1_n2 = 70,
  P_e1_w1 = 71, P_e1_w2 = 72,
  P_enc_temp = 73, P_proj_b = 74, P_proj_w = 75
};

extern "C" void kernel_launch(void* const* d_in, const int* in_sizes, int n_in,
                              void* d_out, int out_size, void* d_ws, size_t ws_size,
                              hipStream_t stream) {
  (void)in_sizes; (void)n_in; (void)out_size; (void)ws_size;
#define PF(idx) ((const float*)d_in[(idx)])
  const float* X  = PF(IN_X);
  const float* T  = PF(IN_T);
  const float* Ty = PF(IN_TY);
  const float* Y  = PF(IN_Y);

  // --------------- workspace arena ---------------
  char* wp = (char*)d_ws;
  auto alloc = [&](size_t bytes) -> void* {
    void* p = (void*)wp;
    wp += (bytes + 255) & ~(size_t)255;
    return p;
  };

  _Float16* wEncConvP = (_Float16*)alloc((size_t)DMOD * 3 * NFEAT * 2);
  _Float16* wDecConvP = (_Float16*)alloc((size_t)DMOD * 3 * NFEAT * 2);
  _Float16* wDistilP  = (_Float16*)alloc((size_t)DMOD * 3 * DMOD * 2);
  _Float16* wP[4][4];                       // [enc0, enc1, decSelf, decCross][q,k,v,o]
  for (int s = 0; s < 4; ++s)
    for (int j = 0; j < 4; ++j)
      wP[s][j] = (_Float16*)alloc((size_t)DMOD * DMOD * 2);
  _Float16* wF1P[3]; _Float16* wF2P[3];     // [enc0, enc1, dec]
  for (int s = 0; s < 3; ++s) {
    wF1P[s] = (_Float16*)alloc((size_t)DMOD * DFFN * 2);
    wF2P[s] = (_Float16*)alloc((size_t)DMOD * DFFN * 2);
  }
  _Float16* wProjP = (_Float16*)alloc((size_t)NFEAT * DMOD * 2);

  const int Menc = BN * LAGL, Menc2 = BN * (LAGL / 2), Mdec = BN * DECL;
  float*    enc  = (float*)alloc((size_t)Menc  * DMOD * 4);
  float*    enc2 = (float*)alloc((size_t)Menc2 * DMOD * 4);
  float*    dec  = (float*)alloc((size_t)Mdec  * DMOD * 4);
  float*    Mid  = (float*)alloc((size_t)Menc  * DFFN * 4);
  _Float16* aH   = (_Float16*)alloc((size_t)Menc * DFFN * 2);
  _Float16* kvH  = (_Float16*)alloc((size_t)Menc * DMOD * 2);
  _Float16* colB = (_Float16*)alloc((size_t)Menc * 3 * DMOD * 2);
  float*    Qf   = (float*)alloc((size_t)Menc * DMOD * 4);   // also distil-conv scratch
  float*    Kf   = (float*)alloc((size_t)Menc * DMOD * 4);
  float*    Vf   = (float*)alloc((size_t)Menc * DMOD * 4);
  float*    Ctx  = (float*)alloc((size_t)Menc * DMOD * 4);
  float*    Upd  = (float*)alloc((size_t)BN * NHEAD * 32 * DHEAD * 4);
  float*    Msc  = (float*)alloc((size_t)BN * NHEAD * LAGL * 4);
  int*      mtop = (int*)alloc((size_t)BN * NHEAD * 32 * 4);

  auto g1d = [](long n) { return dim3((unsigned)((n + 255) / 256)); };

  // --------------- weight packing (per call; deterministic) ---------------
  {
    long n = (long)DMOD * NFEAT * 3;
    packcw_k<<<g1d(n), dim3(256), 0, stream>>>(PF(P_enc_conv), wEncConvP, DMOD, NFEAT);
    packcw_k<<<g1d(n), dim3(256), 0, stream>>>(PF(P_dec_conv), wDecConvP, DMOD, NFEAT);
    long n2 = (long)DMOD * DMOD * 3;
    packcw_k<<<g1d(n2), dim3(256), 0, stream>>>(PF(P_di_w), wDistilP, DMOD, DMOD);
  }
  auto packw = [&](const float* W, _Float16* out, int K, int N) {
    packw_k<<<g1d((long)K * N), dim3(256), 0, stream>>>(W, out, K, N);
  };
  packw(PF(P_e0_wq), wP[0][0], DMOD, DMOD); packw(PF(P_e0_wk), wP[0][1], DMOD, DMOD);
  packw(PF(P_e0_wv), wP[0][2], DMOD, DMOD); packw(PF(P_e0_wo), wP[0][3], DMOD, DMOD);
  packw(PF(P_e1_wq), wP[1][0], DMOD, DMOD); packw(PF(P_e1_wk), wP[1][1], DMOD, DMOD);
  packw(PF(P_e1_wv), wP[1][2], DMOD, DMOD); packw(PF(P_e1_wo), wP[1][3], DMOD, DMOD);
  packw(PF(P_se_wq), wP[2][0], DMOD, DMOD); packw(PF(P_se_wk), wP[2][1], DMOD, DMOD);
  packw(PF(P_se_wv), wP[2][2], DMOD, DMOD); packw(PF(P_se_wo), wP[2][3], DMOD, DMOD);
  packw(PF(P_cr_wq), wP[3][0], DMOD, DMOD); packw(PF(P_cr_wk), wP[3][1], DMOD, DMOD);
  packw(PF(P_cr_wv), wP[3][2], DMOD, DMOD); packw(PF(P_cr_wo), wP[3][3], DMOD, DMOD);
  packw(PF(P_e0_w1), wF1P[0], DMOD, DFFN);  packw(PF(P_e0_w2), wF2P[0], DFFN, DMOD);
  packw(PF(P_e1_w1), wF1P[1], DMOD, DFFN);  packw(PF(P_e1_w2), wF2P[1], DFFN, DMOD);
  packw(PF(P_dl_w1), wF1P[2], DMOD, DFFN);  packw(PF(P_dl_w2), wF2P[2], DFFN, DMOD);
  packw(PF(P_proj_w), wProjP, DMOD, NFEAT);

  // --------------- helpers ---------------
  auto tohalf = [&](const float* x, _Float16* y, long n) {
    tohalf_k<<<g1d(n), dim3(256), 0, stream>>>(x, y, n);
  };
  auto gemm = [&](const _Float16* A, const _Float16* Bt, const float* bias,
                  const float* res, float* C, int M, int N, int K, int act) {
    gemm_wmma_k<<<dim3((unsigned)(M / 64), (unsigned)(N / 128)),
                  dim3(256), 0, stream>>>(A, Bt, bias, res, C, M, N, K, act);
  };
  auto ln = [&](float* x, const float* g, const float* b, int rows) {
    ln_k<<<dim3((unsigned)rows), dim3(256), 0, stream>>>(x, g, b, x);
  };
  auto ufac = [](int L) { int v = 5 * (int)ceil(log((double)L)); return v < L ? v : L; };

  auto run_attn = [&](float* x, const float* xkv, int Lq, int Lk, int causal,
                      const _Float16* const* w, const float* bq, const float* bk,
                      const float* bv, const float* bo) {
    int Mq = BN * Lq, Mk = BN * Lk;
    int U = ufac(Lk), u = ufac(Lq);
    tohalf(x,   aH,  (long)Mq * DMOD);
    tohalf(xkv, kvH, (long)Mk * DMOD);
    gemm(aH,  w[0], bq, nullptr, Qf, Mq, DMOD, DMOD, 0);
    gemm(kvH, w[1], bk, nullptr, Kf, Mk, DMOD, DMOD, 0);
    gemm(kvH, w[2], bv, nullptr, Vf, Mk, DMOD, DMOD, 0);
    attn_m_k<<<g1d((long)BN * NHEAD * Lq), dim3(256), 0, stream>>>(Qf, Kf, Msc, Lq, Lk, U);
    topk_k<<<dim3(BN * NHEAD), dim3(256), 0, stream>>>(Msc, mtop, Lq, u);
    long nc = (long)BN * NHEAD * DHEAD;
    if (causal)
      ctx_cumsum_k<<<g1d(nc), dim3(256), 0, stream>>>(Vf, Ctx, Lk);
    else
      ctx_mean_k<<<g1d(nc), dim3(256), 0, stream>>>(Vf, Ctx, Lk, Lq);
    attn_upd_k<<<dim3(BN * NHEAD * u), dim3(128), 0, stream>>>(Qf, Kf, Vf, mtop, Upd, Lq, Lk, u, causal);
    scatter_k<<<g1d((long)BN * NHEAD * u * DHEAD), dim3(256), 0, stream>>>(Upd, mtop, Ctx, u, Lq);
    tohalf(Ctx, aH, (long)Mq * DMOD);
    gemm(aH, w[3], bo, /*residual=*/x, x, Mq, DMOD, DMOD, 0);   // in-place x += attn
  };

  auto run_ffn = [&](float* x, int M, const _Float16* w1p, const float* b1,
                     const _Float16* w2p, const float* b2) {
    tohalf(x, aH, (long)M * DMOD);
    gemm(aH, w1p, b1, nullptr, Mid, M, DFFN, DMOD, /*gelu*/1);
    tohalf(Mid, aH, (long)M * DFFN);
    gemm(aH, w2p, b2, /*residual=*/x, x, M, DMOD, DFFN, 0);
  };

  // --------------- encoder embedding ---------------
  im2col_k<<<g1d((long)Menc * 3 * NFEAT), dim3(256), 0, stream>>>(
      X, colB, BN, LAGL, NFEAT, (long)NFEAT * LAGL, 1L, (long)LAGL, LAGL);
  gemm(colB, wEncConvP, nullptr, nullptr, enc, Menc, DMOD, 3 * NFEAT, 0);
  embed_k<<<g1d((long)Menc * DMOD), dim3(256), 0, stream>>>(enc, T, PF(P_enc_temp), LAGL);

  // --------------- encoder layer 0 (L=336) ---------------
  run_attn(enc, enc, LAGL, LAGL, 0, wP[0], PF(P_e0_bq), PF(P_e0_bk), PF(P_e0_bv), PF(P_e0_bo));
  ln(enc, PF(P_e0_g1), PF(P_e0_n1), Menc);
  run_ffn(enc, Menc, wF1P[0], PF(P_e0_b1), wF2P[0], PF(P_e0_b2));
  ln(enc, PF(P_e0_g2), PF(P_e0_n2), Menc);

  // --------------- distil conv + ELU + maxpool (336 -> 168) ---------------
  im2col_k<<<g1d((long)Menc * 3 * DMOD), dim3(256), 0, stream>>>(
      enc, colB, BN, LAGL, DMOD, (long)LAGL * DMOD, (long)DMOD, 1L, LAGL);
  gemm(colB, wDistilP, PF(P_di_b), nullptr, Qf, Menc, DMOD, 3 * DMOD, 0);
  bnelu_k<<<g1d((long)Menc * DMOD), dim3(256), 0, stream>>>(Qf, PF(P_di_gamma), PF(P_di_beta),
                                                            (long)Menc * DMOD);
  maxpool_k<<<g1d((long)Menc2 * DMOD), dim3(256), 0, stream>>>(Qf, enc2, LAGL);

  // --------------- encoder layer 1 (L=168) + final LN ---------------
  run_attn(enc2, enc2, LAGL / 2, LAGL / 2, 0, wP[1], PF(P_e1_bq), PF(P_e1_bk), PF(P_e1_bv), PF(P_e1_bo));
  ln(enc2, PF(P_e1_g1), PF(P_e1_n1), Menc2);
  run_ffn(enc2, Menc2, wF1P[1], PF(P_e1_b1), wF2P[1], PF(P_e1_b2));
  ln(enc2, PF(P_e1_g2), PF(P_e1_n2), Menc2);
  ln(enc2, PF(P_enc_g), PF(P_enc_b), Menc2);

  // --------------- decoder embedding (first 48 steps of Y, rest zero) ---------------
  im2col_k<<<g1d((long)Mdec * 3 * NFEAT), dim3(256), 0, stream>>>(
      Y, colB, BN, DECL, NFEAT, (long)NFEAT * DECL, 1L, (long)DECL, LABL);
  gemm(colB, wDecConvP, nullptr, nullptr, dec, Mdec, DMOD, 3 * NFEAT, 0);
  embed_k<<<g1d((long)Mdec * DMOD), dim3(256), 0, stream>>>(dec, Ty, PF(P_dec_temp), DECL);

  // --------------- decoder layer ---------------
  run_attn(dec, dec, DECL, DECL, /*causal*/1, wP[2], PF(P_se_bq), PF(P_se_bk), PF(P_se_bv), PF(P_se_bo));
  ln(dec, PF(P_dl_g1), PF(P_dl_n1), Mdec);
  run_attn(dec, enc2, DECL, LAGL / 2, 0, wP[3], PF(P_cr_bq), PF(P_cr_bk), PF(P_cr_bv), PF(P_cr_bo));
  ln(dec, PF(P_dl_g2), PF(P_dl_n2), Mdec);
  run_ffn(dec, Mdec, wF1P[2], PF(P_dl_b1), wF2P[2], PF(P_dl_b2));
  ln(dec, PF(P_dl_g3), PF(P_dl_n3), Mdec);
  ln(dec, PF(P_dec_g), PF(P_dec_b), Mdec);

  // --------------- projection + output transpose ---------------
  tohalf(dec, aH, (long)Mdec * DMOD);
  gemm(aH, wProjP, PF(P_proj_b), nullptr, Kf, Mdec, NFEAT, DMOD, 0);
  final_k<<<g1d((long)BN * NFEAT * PREDL), dim3(256), 0, stream>>>(Kf, (float*)d_out);
#undef PF
}
